// EquivariantProductBasisLayer_29850022707550
// MI455X (gfx1250) — matrix operand order repaired and checked by reading, not compiled, not run on hardware
//
#include <hip/hip_runtime.h>

// CDNA5 (gfx1250) MACE EquivariantProductBasis layer.
// Strategy: rewrite the symmetric contraction as GEMMs over monomials of x
// with generated A (x_a*x_b*x_j) and pre-swizzled f16 B (the u-tensors),
// accumulate fp32 via v_wmma_f32_16x16x32_f16, keep all intermediates in LDS.
// B slabs are staged workgroup-wide via global_load_async_to_lds_b128
// (ASYNCcnt) with double buffering, so the L2-resident 0.9MB B matrix is read
// once per WG per k-slab instead of once per wave.

typedef _Float16 v16h __attribute__((ext_vector_type(16)));
typedef float    v8f  __attribute__((ext_vector_type(8)));

#define IDIV(i,a,b,c,d) ((i)==0?(a):(i)==1?(b):(i)==2?(c):(d))

struct UPtrs {
  const float* u1[4];
  const float* u2[4];
  const float* u3[4];
};

static constexpr int S3 = 128*7*32*16;   // order-3 swizzled B: 458752 halfs (K=4096 x 112 cols)
static constexpr int S2 = 8*3*32*16;     // order-2: 12288 halfs (K=256 x 48 cols)
static constexpr int S1 = 32*16;         // order-1: 512 halfs  (K=16(pad32) x 16 cols)

// ---------------------------------------------------------------------------
// Prep: pack u3/u2/u1 (fp32) into f16 WMMA B-fragment layout:
//   ws[kstep][coltile][lane(32)][elem(16)], elem e of lane L = B[K][col] with
//   K = kstep*32 + (e>>3)*16 + (L>>4)*8 + (e&7), col = coltile*16 + (L&15).
// ---------------------------------------------------------------------------
__global__ __launch_bounds__(256) void prep_uswizzle(UPtrs U, _Float16* ws3,
                                                     _Float16* ws2, _Float16* ws1)
{
  const int idx = blockIdx.x * 256 + threadIdx.x;
  if (idx < S3) {
    const int e = idx & 15, L = (idx >> 4) & 31;
    const int rest = idx >> 9;
    const int tile = rest % 7, kstep = rest / 7;
    const int half = L >> 4, c = L & 15;
    const int K = kstep*32 + (e>>3)*16 + half*8 + (e&7);          // 0..4095 -> (a,b,j)
    const int col = tile*16 + c;                   // 0..111  -> (irrep,kpath,ii)
    const int a = K >> 8, b = (K >> 4) & 15, j = K & 15;
    int i, rem;
    if (col < 7)       { i = 0; rem = col;      }
    else if (col < 28) { i = 1; rem = col - 7;  }
    else if (col < 63) { i = 2; rem = col - 28; }
    else               { i = 3; rem = col - 63; }
    const int di = IDIV(i,1,3,5,7);
    const int kp = rem / di, ii = rem % di;
    // u3_i shape (16,16,16,7,di)
    const float v = U.u3[i][(((a*16 + b)*16 + j)*7 + kp)*di + ii];
    ws3[idx] = (_Float16)v;
  } else if (idx < S3 + S2) {
    const int id2 = idx - S3;
    const int e = id2 & 15, L = (id2 >> 4) & 31;
    const int rest = id2 >> 9;
    const int tile = rest % 3, kstep = rest / 3;
    const int half = L >> 4, c = L & 15;
    const int K = kstep*32 + (e>>3)*16 + half*8 + (e&7);          // 0..255 -> (a,b)
    const int col = tile*16 + c;                   // 0..47
    const int a = K >> 4, b = K & 15;
    int i, rem;
    if (col < 3)       { i = 0; rem = col;      }
    else if (col < 12) { i = 1; rem = col - 3;  }
    else if (col < 27) { i = 2; rem = col - 12; }
    else               { i = 3; rem = col - 27; }
    const int di = IDIV(i,1,3,5,7);
    const int kp = rem / di, ii = rem % di;
    // u2_i shape (16,16,3,di)
    const float v = U.u2[i][((a*16 + b)*3 + kp)*di + ii];
    ws2[id2] = (_Float16)v;
  } else if (idx < S3 + S2 + S1) {
    const int id1 = idx - S3 - S2;
    const int e = id1 & 15, L = id1 >> 4;
    const int half = L >> 4, c = L & 15;
    const int K = (e>>3)*16 + half*8 + (e&7);                     // 0..31, rows >=16 are zero pad
    float v = 0.0f;
    if (K < 16) {
      int i, ii;
      if (c < 1)      { i = 0; ii = c;     }
      else if (c < 4) { i = 1; ii = c - 1; }
      else if (c < 9) { i = 2; ii = c - 4; }
      else            { i = 3; ii = c - 9; }
      const int di = IDIV(i,1,3,5,7);
      v = U.u1[i][K*di + ii];                      // u1_i shape (16,1,di)
    }
    ws1[id1] = (_Float16)v;
  }
}

// ---------------------------------------------------------------------------
// Main: one workgroup (8 waves) per node. Wave w owns channels w*16..w*16+15
// as the 16 GEMM rows; 11 col-tiles (7 order-3, 3 order-2, 1 order-1).
// Order-3 B slabs (2 ksteps = 14336B) are async-staged into LDS, double
// buffered, shared by all 8 waves.
// ---------------------------------------------------------------------------
__global__ __launch_bounds__(256) void mace_product_basis(
    const float* __restrict__ x,        // (512,128,16)
    const int*   __restrict__ specie,   // (512,)
    const float* __restrict__ w1,       // (4,10,1,128)
    const float* __restrict__ w2,       // (4,10,3,128)
    const float* __restrict__ w3,       // (4,10,7,128)
    const float* __restrict__ wlin,     // (4,128,128)
    const _Float16* __restrict__ B3,
    const _Float16* __restrict__ B2,
    const _Float16* __restrict__ B1,
    float* __restrict__ out)            // (512,2048)
{
  extern __shared__ float smem[];
  float*     Xs     = smem;                       // 128*16  = 2048 f
  float*     Plds   = smem + 2048;                // 128*176 = 22528 f
  float*     Zlds   = smem + 2048 + 22528;        // 128*16  = 2048 f
  _Float16*  Bstage = (_Float16*)(smem + 26624);  // 2 x 7168 halfs (2 x 14336 B)

  const int tid  = (int)threadIdx.x;
  const int n    = (int)blockIdx.x;
  const int w    = tid >> 5;
  const int lane = tid & 31;
  const int half = lane >> 4;
  const int r    = lane & 15;
  const int row  = w*16 + r;            // channel index within node

  // Raw 32-bit LDS byte address of the staging area (generic LDS-aperture
  // addresses carry the LDS offset in addr[31:0]).
  const uint32_t ldsB = (uint32_t)(uintptr_t)Bstage;

  // Async-stage one 2-kstep slab (28 x 512B) of B3 into LDS buffer bi.
  // Waves 0..6 issue 4 per-lane b128 copies each (wave covers 512B/instr).
  auto issue_slab = [&](int s, int bi) {
    if (w < 7) {
      const char*    g = (const char*)B3 + (size_t)s*14336 + (size_t)(w*4)*512
                                         + (size_t)lane*16;
      const uint32_t l = ldsB + (uint32_t)bi*14336u + (uint32_t)(w*4)*512u
                              + (uint32_t)lane*16u;
      #pragma unroll
      for (int q = 0; q < 4; ++q) {
        asm volatile("global_load_async_to_lds_b128 %0, %1, off"
                     :: "v"(l + (uint32_t)q*512u), "v"(g + q*512) : "memory");
      }
    }
  };

  issue_slab(0, 0);   // prefetch first slab; completion checked in the loop

  // Phase 0: stage this node's x rows (128 x 16 fp32) into LDS
  {
    const int rr = tid >> 1;
    const int o  = (tid & 1) * 8;
    const float4* s4 = (const float4*)(x + ((size_t)n*128 + rr)*16 + o);
    const float4 v0 = s4[0], v1 = s4[1];
    *(float4*)&Xs[rr*16 + o]     = v0;
    *(float4*)&Xs[rr*16 + o + 4] = v1;
  }
  __syncthreads();

  // A-layout: lane (r,half) needs x[row][half*8 + q], q=0..7 -> registers
  float xj[8];
  {
    const float4 q0 = *(const float4*)&Xs[row*16 + half*8];
    const float4 q1 = *(const float4*)&Xs[row*16 + half*8 + 4];
    xj[0]=q0.x; xj[1]=q0.y; xj[2]=q0.z; xj[3]=q0.w;
    xj[4]=q1.x; xj[5]=q1.y; xj[6]=q1.z; xj[7]=q1.w;
  }

  v8f acc[11] = {};   // 7 order-3 + 3 order-2 + 1 order-1 col-tiles (fp32)

  // ---- order 3: K=4096 (m = a*256+b*16+j, A = x_a*x_b*x_j), cols 0..111 ----
  for (int s = 0; s < 64; ++s) {
    const int bi = s & 1;
    asm volatile("s_wait_asynccnt 0x0" ::: "memory");
    __syncthreads();                     // slab bi resident; bi^1 fully consumed
    if (s + 1 < 64) issue_slab(s + 1, bi ^ 1);

    const _Float16* bufl = Bstage + bi*7168 + lane*16;
    #pragma unroll
    for (int ksl = 0; ksl < 2; ++ksl) {
      const int ks = s*2 + ksl;
      const float  xa = Xs[row*16 + (ks >> 3)];
      const float2 xb = *(const float2*)&Xs[row*16 + ((ks & 7) << 1)];
      const float p0 = xa * xb.x;        // k_local 0..15  share a,b0
      const float p1 = xa * xb.y;        // k_local 16..31 share a,b1
      v16h a;
      #pragma unroll
      for (int q = 0; q < 8; ++q) {
        a[q]     = (_Float16)(p0 * xj[q]);   // K = half*8+q
        a[q + 8] = (_Float16)(p1 * xj[q]);   // K = 16+half*8+q
      }
      const _Float16* bb = bufl + ksl*3584;  // 7*512 halfs per kstep
      #pragma unroll
      for (int t = 0; t < 7; ++t) {
        const v16h b = *(const v16h*)(bb + t*512);
        acc[t] = __builtin_amdgcn_wmma_f32_16x16x32_f16(
                     false, a, false, b, (short)0, acc[t], false, false);
      }
    }
  }

  // ---- order 2: K=256 (m = a*16+b, A = x_a*x_b), cols 112..159 ----
  const _Float16* b2l = B2 + lane*16;
  #pragma unroll
  for (int ks = 0; ks < 8; ++ks) {
    const float2 pp = *(const float2*)&Xs[row*16 + ks*2];
    v16h a;
    #pragma unroll
    for (int q = 0; q < 8; ++q) {
      a[q]     = (_Float16)(pp.x * xj[q]);
      a[q + 8] = (_Float16)(pp.y * xj[q]);
    }
    const _Float16* bb = b2l + ks*3*512;
    #pragma unroll
    for (int t = 0; t < 3; ++t) {
      const v16h b = *(const v16h*)(bb + t*512);
      acc[7 + t] = __builtin_amdgcn_wmma_f32_16x16x32_f16(
                       false, a, false, b, (short)0, acc[7 + t], false, false);
    }
  }

  // ---- order 1: K=16 (zero-padded to 32), cols 160..175 ----
  {
    v16h a;
    #pragma unroll
    for (int q = 0; q < 8; ++q) {
      a[q]     = (_Float16)xj[q];          // K = half*8+q < 16
      a[q + 8] = (_Float16)0.0f;           // K >= 16 pad
    }
    const v16h b = *(const v16h*)(B1 + lane*16);
    acc[10] = __builtin_amdgcn_wmma_f32_16x16x32_f16(
                  false, a, false, b, (short)0, acc[10], false, false);
  }

  // ---- store P (c x 176) to LDS; D VGPR vg -> row half*8+vg, col = r ----
  #pragma unroll
  for (int t = 0; t < 11; ++t) {
    #pragma unroll
    for (int vg = 0; vg < 8; ++vg) {
      Plds[(w*16 + half*8 + vg)*176 + t*16 + r] = acc[t][vg];
    }
  }
  __syncthreads();

  // ---- species-conditioned path weighting: Z[c][iicat] ----
  {
    const int c = tid & 127;
    const int g = tid >> 7;
    const int s = specie[n];
    const float* prow = &Plds[c*176];
    #pragma unroll
    for (int q = 0; q < 8; ++q) {
      const int iicat = g*8 + q;          // 0..15 over concat(di)
      int i, ii;
      if (iicat < 1)      { i = 0; ii = 0;          }
      else if (iicat < 4) { i = 1; ii = iicat - 1;  }
      else if (iicat < 9) { i = 2; ii = iicat - 4;  }
      else                { i = 3; ii = iicat - 9;  }
      const int di = IDIV(i,1,3,5,7);
      const int o3 = IDIV(i,0,7,28,63);
      const int o2 = IDIV(i,0,3,12,27);
      const int o1 = IDIV(i,0,1,4,9);
      float z = 0.0f;
      #pragma unroll
      for (int k = 0; k < 7; ++k)
        z += w3[((i*10 + s)*7 + k)*128 + c] * prow[o3 + k*di + ii];
      #pragma unroll
      for (int k = 0; k < 3; ++k)
        z += w2[((i*10 + s)*3 + k)*128 + c] * prow[112 + o2 + k*di + ii];
      z += w1[(i*10 + s)*128 + c] * prow[160 + o1 + ii];
      Zlds[c*16 + iicat] = z;
    }
  }
  __syncthreads();

  // ---- per-irrep e3nn Linear: y[n,f,ii] = (1/sqrt(C)) * sum_c Z[c,ii]*wlin[i][c][f] ----
  {
    const int f = tid & 127;
    const int g = tid >> 7;
    float* o = out + (size_t)n * 2048;
    constexpr float INV = 0.08838834764831845f;   // 1/sqrt(128)
    float av[8] = {};
    if (g == 0) {   // iicat 0..7: irreps {0, 1,1,1, 2,2,2,2}
      for (int c = 0; c < 128; ++c) {
        const float4 z0 = *(const float4*)&Zlds[c*16];
        const float4 z1 = *(const float4*)&Zlds[c*16 + 4];
        const float wl0 = wlin[c*128 + f];
        const float wl1 = wlin[(128 + c)*128 + f];
        const float wl2 = wlin[(256 + c)*128 + f];
        av[0] += z0.x*wl0;
        av[1] += z0.y*wl1; av[2] += z0.z*wl1; av[3] += z0.w*wl1;
        av[4] += z1.x*wl2; av[5] += z1.y*wl2; av[6] += z1.z*wl2; av[7] += z1.w*wl2;
      }
      o[f]              = av[0]*INV;
      o[128 + f*3 + 0]  = av[1]*INV;
      o[128 + f*3 + 1]  = av[2]*INV;
      o[128 + f*3 + 2]  = av[3]*INV;
      o[512 + f*5 + 0]  = av[4]*INV;
      o[512 + f*5 + 1]  = av[5]*INV;
      o[512 + f*5 + 2]  = av[6]*INV;
      o[512 + f*5 + 3]  = av[7]*INV;
    } else {        // iicat 8..15: irreps {2(ii=4), 3 x7}
      for (int c = 0; c < 128; ++c) {
        const float4 z0 = *(const float4*)&Zlds[c*16 + 8];
        const float4 z1 = *(const float4*)&Zlds[c*16 + 12];
        const float wl2 = wlin[(256 + c)*128 + f];
        const float wl3 = wlin[(384 + c)*128 + f];
        av[0] += z0.x*wl2;
        av[1] += z0.y*wl3; av[2] += z0.z*wl3; av[3] += z0.w*wl3;
        av[4] += z1.x*wl3; av[5] += z1.y*wl3; av[6] += z1.z*wl3; av[7] += z1.w*wl3;
      }
      o[512  + f*5 + 4] = av[0]*INV;
      o[1152 + f*7 + 0] = av[1]*INV;
      o[1152 + f*7 + 1] = av[2]*INV;
      o[1152 + f*7 + 2] = av[3]*INV;
      o[1152 + f*7 + 3] = av[4]*INV;
      o[1152 + f*7 + 4] = av[5]*INV;
      o[1152 + f*7 + 5] = av[6]*INV;
      o[1152 + f*7 + 6] = av[7]*INV;
    }
  }
}

extern "C" void kernel_launch(void* const* d_in, const int* in_sizes, int n_in,
                              void* d_out, int out_size, void* d_ws, size_t ws_size,
                              hipStream_t stream) {
  (void)in_sizes; (void)n_in; (void)out_size; (void)ws_size;
  const float* x  = (const float*)d_in[0];
  const int*   sp = (const int*)d_in[1];
  UPtrs U;
  for (int i = 0; i < 4; ++i) {
    U.u1[i] = (const float*)d_in[2 + 3*i];
    U.u2[i] = (const float*)d_in[3 + 3*i];
    U.u3[i] = (const float*)d_in[4 + 3*i];
  }
  const float* w1   = (const float*)d_in[14];
  const float* w2   = (const float*)d_in[15];
  const float* w3   = (const float*)d_in[16];
  const float* wlin = (const float*)d_in[17];

  _Float16* ws3 = (_Float16*)d_ws;        // 458752 halfs
  _Float16* ws2 = ws3 + S3;               // 12288 halfs
  _Float16* ws1 = ws2 + S2;               // 512 halfs

  prep_uswizzle<<<(S3 + S2 + S1) / 256, 256, 0, stream>>>(U, ws3, ws2, ws1);

  // Xs(8KB) + P(88KB) + Z(8KB) + B double-buffer(28KB) = 135168 B
  const size_t shmem = (26624 + 7168) * sizeof(float);
  hipFuncSetAttribute((const void*)mace_product_basis,
                      hipFuncAttributeMaxDynamicSharedMemorySize, (int)shmem);
  mace_product_basis<<<512, 256, shmem, stream>>>(x, sp, w1, w2, w3, wlin,
                                                  ws3, ws2, ws1, (float*)d_out);
}
